// get_model_11364483465264
// MI455X (gfx1250) — compile-verified, matrix operand order checked by reading, not run on hardware
//
#include <hip/hip_runtime.h>

typedef __attribute__((ext_vector_type(16))) _Float16 v16h;
typedef __attribute__((ext_vector_type(8)))  float    v8f;
typedef __attribute__((ext_vector_type(4)))  float    v4f;
typedef unsigned int u32x4 __attribute__((ext_vector_type(4)));
typedef int          i32x8 __attribute__((ext_vector_type(8)));
typedef int          i32x4 __attribute__((ext_vector_type(4)));

#define BN_SCALE 0.99999500003749972f   /* 1/sqrt(1+1e-5) */
#define NPTS  16384
#define BATCH 2
#define S8    8

// ---------------------------------------------------------------- FPS
__global__ void fps_kernel(const float* __restrict__ xyz, int* __restrict__ fps_idx,
                           float* __restrict__ dist) {
  const int b = blockIdx.x, t = threadIdx.x;
  const float* X = xyz + (size_t)b * 1027 * NPTS;
  float* D = dist + (size_t)b * NPTS;
  __shared__ float sval[256];
  __shared__ int   sidx[256];
  __shared__ int   sfar;
  for (int n = t; n < NPTS; n += 256) D[n] = 1e10f;
  if (t == 0) sfar = 0;
  __syncthreads();
  for (int it = 0; it < S8; ++it) {
    int far = sfar;
    if (t == 0) fps_idx[b * S8 + it] = far;
    float cx = X[far], cy = X[NPTS + far], cz = X[2 * NPTS + far];
    float best = -1.0f; int bi = 0x7fffffff;
    for (int n = t; n < NPTS; n += 256) {
      float dx = X[n] - cx, dy = X[NPTS + n] - cy, dz = X[2 * NPTS + n] - cz;
      float d = fminf(D[n], dx*dx + dy*dy + dz*dz);
      D[n] = d;
      if (d > best) { best = d; bi = n; }
    }
    sval[t] = best; sidx[t] = bi;
    __syncthreads();
    for (int stp = 128; stp > 0; stp >>= 1) {
      if (t < stp) {
        if (sval[t+stp] > sval[t] ||
            (sval[t+stp] == sval[t] && sidx[t+stp] < sidx[t])) {
          sval[t] = sval[t+stp]; sidx[t] = sidx[t+stp];
        }
      }
      __syncthreads();
    }
    if (t == 0) sfar = sidx[0];
    __syncthreads();
  }
}

__global__ void gather_new_xyz(const float* __restrict__ xyz,
                               const int* __restrict__ fps_idx,
                               float* __restrict__ new_xyz) {
  int i = threadIdx.x;
  if (i >= BATCH * S8 * 3) return;
  int b = i / (S8 * 3), r = i % (S8 * 3), s = r / 3, c = r % 3;
  int n = fps_idx[b * S8 + s];
  new_xyz[i] = xyz[((size_t)b * 1027 + c) * NPTS + n];
}

// ---------------------------------------------------------------- ball query
__global__ void ball_query(const float* __restrict__ xyz,
                           const float* __restrict__ new_xyz,
                           int* __restrict__ gi, float r2, int K) {
  int t = threadIdx.x;
  if (t >= BATCH * S8) return;
  int b = t >> 3;
  const float* X = xyz + (size_t)b * 1027 * NPTS;
  float cx = new_xyz[t*3], cy = new_xyz[t*3+1], cz = new_xyz[t*3+2];
  int cnt = 0, first = 0;
  for (int n = 0; n < NPTS && cnt < K; ++n) {
    float dx = X[n]-cx, dy = X[NPTS+n]-cy, dz = X[2*NPTS+n]-cz;
    if (dx*dx + dy*dy + dz*dz <= r2) {
      if (cnt == 0) first = n;
      gi[t*K + cnt] = n; ++cnt;
    }
  }
  for (; cnt < K; ++cnt) gi[t*K + cnt] = first;
}

// ---------------------------------------------------------------- grouped feats
__global__ void build_gp(const float* __restrict__ xyz,
                         const float* __restrict__ new_xyz,
                         const int* __restrict__ gi,
                         float* __restrict__ gp, int K) {
  int i = blockIdx.x * 256 + threadIdx.x;
  int R = BATCH * S8 * K;
  if (i >= R * 1030) return;
  int r = i / 1030, c = i % 1030;
  int bs = r / K, k = r % K;
  int b = bs >> 3;
  int n = gi[bs*K + k];
  const float* X = xyz + (size_t)b * 1027 * NPTS;
  float v;
  if (c < 1027) v = X[(size_t)c * NPTS + n];
  else { int cc = c - 1027; v = X[(size_t)cc * NPTS + n] - new_xyz[bs*3 + cc]; }
  gp[(size_t)r * 1030 + c] = v;
}

// ---------------------------------------------------------------- scalar MLP (tiny stages)
__global__ void mlp_scalar(const float* __restrict__ X, const float* __restrict__ W,
                           const float* __restrict__ b, const float* __restrict__ g,
                           const float* __restrict__ be, float* __restrict__ Y,
                           int R, int Cin, int Cout, int bn) {
  int i = blockIdx.x * 256 + threadIdx.x;
  if (i >= R * Cout) return;
  int r = i / Cout, o = i % Cout;
  const float* x = X + (size_t)r * Cin;
  const float* w = W + (size_t)o * Cin;
  float acc = 0.0f;
  for (int c = 0; c < Cin; ++c) acc = fmaf(x[c], w[c], acc);
  acc += b[o];
  if (bn) acc = fmaxf(0.0f, g[o] * (acc * BN_SCALE) + be[o]);
  Y[i] = acc;
}

__global__ void max_over_k(const float* __restrict__ H, float* __restrict__ l1_pts,
                           int K, int C, int coff) {
  int i = blockIdx.x * 256 + threadIdx.x;
  if (i >= BATCH * S8 * C) return;
  int bs = i / C, c = i % C;
  float m = -3.4e38f;
  for (int k = 0; k < K; ++k) m = fmaxf(m, H[(size_t)(bs*K + k) * C + c]);
  l1_pts[(size_t)bs * 1024 + coff + c] = m;
}

__global__ void build_sa3_in(const float* __restrict__ new_xyz,
                             const float* __restrict__ l1_pts,
                             float* __restrict__ out) {
  int i = blockIdx.x * 256 + threadIdx.x;
  if (i >= 16 * 1027) return;
  int r = i / 1027, c = i % 1027;
  out[i] = (c < 3) ? new_xyz[r*3 + c] : l1_pts[r*1024 + (c-3)];
}

__global__ void max_over_s(const float* __restrict__ H, float* __restrict__ out) {
  int i = blockIdx.x * 256 + threadIdx.x;
  if (i >= BATCH * 1024) return;
  int b = i / 1024, c = i % 1024;
  float m = -3.4e38f;
  for (int s = 0; s < 8; ++s) m = fmaxf(m, H[(b*8 + s)*1024 + c]);
  out[i] = m;
}

__global__ void build_fp3_in(const float* __restrict__ l1_pts,
                             const float* __restrict__ l3max,
                             float* __restrict__ out) {
  int i = blockIdx.x * 256 + threadIdx.x;
  if (i >= 16 * 2048) return;
  int r = i / 2048, c = i % 2048;
  out[i] = (c < 1024) ? l1_pts[r*1024 + c] : l3max[(r >> 3)*1024 + (c-1024)];
}

// ---------------------------------------------------------------- 3-NN interpolation weights
__global__ void knn3_weights(const float* __restrict__ xyz,
                             const float* __restrict__ new_xyz,
                             int* __restrict__ idx3, float* __restrict__ w3) {
  int p = blockIdx.x * 256 + threadIdx.x;
  if (p >= BATCH * NPTS) return;
  int b = p / NPTS, n = p % NPTS;
  const float* X = xyz + (size_t)b * 1027 * NPTS;
  float px = X[n], py = X[NPTS + n], pz = X[2*NPTS + n];
  float d[8];
  #pragma unroll
  for (int s = 0; s < 8; ++s) {
    float dx = px - new_xyz[(b*8+s)*3];
    float dy = py - new_xyz[(b*8+s)*3+1];
    float dz = pz - new_xyz[(b*8+s)*3+2];
    d[s] = dx*dx + dy*dy + dz*dz;
  }
  int used0 = -1, used1 = -1;
  int id[3]; float w[3]; float wsum = 0.0f;
  for (int j = 0; j < 3; ++j) {
    float bestd = 3.4e38f; int bi = 0;
    for (int s = 0; s < 8; ++s) {
      if (s == used0 || s == used1) continue;
      if (d[s] < bestd) { bestd = d[s]; bi = s; }
    }
    if (j == 0) used0 = bi; else if (j == 1) used1 = bi;
    id[j] = bi; w[j] = 1.0f / (bestd + 1e-8f); wsum += w[j];
  }
  for (int j = 0; j < 3; ++j) { idx3[p*3+j] = id[j]; w3[p*3+j] = w[j] / wsum; }
}

// ---------------------------------------------------------------- fp1 input (f16, fused interp)
__global__ void build_a_kernel(const float* __restrict__ xyz,
                               const int* __restrict__ idx3,
                               const float* __restrict__ w3,
                               const float* __restrict__ fp3o,
                               _Float16* __restrict__ A) {
  size_t i = (size_t)blockIdx.x * 256 + threadIdx.x;
  const size_t total = (size_t)BATCH * NPTS * 2080;
  if (i >= total) return;
  int p = (int)(i / 2080), k = (int)(i % 2080);
  int b = p >> 14, n = p & (NPTS - 1);
  float v = 0.0f;
  if (k < 3)        v = xyz[((size_t)b*1027 + k) * NPTS + n];
  else if (k < 1030) v = xyz[((size_t)b*1027 + (k-3)) * NPTS + n];
  else if (k < 2054) {
    int c = k - 1030;
    v = w3[p*3+0] * fp3o[(b*8 + idx3[p*3+0])*1024 + c]
      + w3[p*3+1] * fp3o[(b*8 + idx3[p*3+1])*1024 + c]
      + w3[p*3+2] * fp3o[(b*8 + idx3[p*3+2])*1024 + c];
  }
  A[i] = (_Float16)v;
}

__global__ void conv_weight(const float* __restrict__ W, _Float16* __restrict__ Wf,
                            int NoutReal, int NoutPad, int Kin, int Kpad) {
  int i = blockIdx.x * 256 + threadIdx.x;
  if (i >= NoutPad * Kpad) return;
  int n = i / Kpad, k = i % Kpad;
  float v = (n < NoutReal && k < Kin) ? W[(size_t)n * Kin + k] : 0.0f;
  Wf[i] = (_Float16)v;
}

// ---------------------------------------------------------------- TDM 2D tile load
// D# per CDNA5 ISA 8.3/8.4: group0 {count=1, lds_addr, global_addr, type=2},
// group1 {data_size=2B, tensor_dim0=kpad, tensor_dim1=M, tile=32 x rows,
//         tensor_dim0_stride=kpad}; groups 2/3 zero (2D tensor).
// Toolchain here carries the 6-arg builtin: (g0, g1, g2, g3, g4, cpol).
__device__ inline void tdm_load_tile(const _Float16* gsrc, unsigned lds_off,
                                     int kpad, int rows) {
  unsigned long ga = (unsigned long)(size_t)gsrc;
  u32x4 g0;
  g0[0] = 1u;                                            // count=1, user mode
  g0[1] = lds_off;                                       // LDS byte address
  g0[2] = (unsigned)(ga & 0xffffffffu);                  // global_addr[31:0]
  g0[3] = (unsigned)((ga >> 32) & 0x1ffffffu) | (2u << 30); // addr[56:32] | type=2
  i32x8 g1;
  g1[0] = (int)(1u << 16);                               // data_size=1 (2 bytes)
  g1[1] = (int)(((unsigned)kpad & 0xffffu) << 16);       // tensor_dim0[15:0]
  g1[2] = (int)((((unsigned)kpad >> 16) & 0xffffu) |
                ((32768u & 0xffffu) << 16));             // dim0[31:16] | dim1[15:0]
  g1[3] = (int)(((32768u >> 16) & 0xffffu) | (32u << 16)); // dim1[31:16] | tile_dim0=32
  g1[4] = (int)((unsigned)rows & 0xffffu);               // tile_dim1 = rows, tile_dim2=0
  g1[5] = kpad;                                          // tensor_dim0_stride[31:0]
  g1[6] = 0;                                             // stride0[47:32] | stride1[15:0]
  g1[7] = 0;
  i32x4 z4 = {0, 0, 0, 0};
  i32x8 z8 = {0, 0, 0, 0, 0, 0, 0, 0};
  __builtin_amdgcn_tensor_load_to_lds(g0, g1, z4, z4, z8, 0);
}

// ---------------------------------------------------------------- WMMA GEMM
// Block: 256 threads = 8 waves. Tile 256(M) x 64(N), BK = 32.
// Wave w owns rows [32w,32w+32): 2 A-frags x 4 N-tiles = 8 WMMAs per K step.
// Staging via Tensor Data Mover (wave 0 issues, TENSORcnt + barrier sync).
__global__ __launch_bounds__(256) void gemm_wmma(
    const _Float16* __restrict__ A, const _Float16* __restrict__ Wf,
    const float* __restrict__ bias, const float* __restrict__ gam,
    const float* __restrict__ bet,
    _Float16* __restrict__ Ybf, float* __restrict__ Yt,
    int Kpad, int NoutReal, int NoutPad, int bnrelu) {
  __shared__ _Float16 sA[256 * 32];   // 16 KB
  __shared__ _Float16 sB[64 * 32];    //  4 KB
  const int t = threadIdx.x;
  const int wv = t >> 5, lane = t & 31;
  const int lr = lane & 15, half = lane >> 4;
  const int m0 = blockIdx.x * 256;
  const int n0 = blockIdx.y * 64;

  v8f zero = {0.f,0.f,0.f,0.f,0.f,0.f,0.f,0.f};
  v8f acc[2][4];
  #pragma unroll
  for (int mi = 0; mi < 2; ++mi)
    #pragma unroll
    for (int j = 0; j < 4; ++j) acc[mi][j] = zero;

  const unsigned ldsA = (unsigned)(size_t)&sA[0];  // LDS aperture: low 32 bits = LDS offset
  const unsigned ldsB = (unsigned)(size_t)&sB[0];

  union FU { v16h v; v4f f[2]; };

  for (int k0 = 0; k0 < Kpad; k0 += 32) {
    __syncthreads();                 // all waves done reading previous LDS tiles
    if (t < 32) {                    // wave 0 drives the TDM
      tdm_load_tile(A  + (size_t)m0 * Kpad + k0, ldsA, Kpad, 256);
      tdm_load_tile(Wf + (size_t)n0 * Kpad + k0, ldsB, Kpad, 64);
      __builtin_amdgcn_s_wait_tensorcnt(0);
    }
    __syncthreads();                 // release consumers

    // A fragments: ISA 16-bit 16x32 layout (lanes 0-15: K0-7,K16-23; 16-31: K8-15,K24-31)
    FU af[2];
    #pragma unroll
    for (int mi = 0; mi < 2; ++mi) {
      const v4f* pa = (const v4f*)(&sA[(32*wv + 16*mi + lr) * 32]);
      af[mi].f[0] = pa[half];
      af[mi].f[1] = pa[2 + half];
    }
    #pragma unroll
    for (int j = 0; j < 4; ++j) {
      FU bf;
      const v4f* pb = (const v4f*)(&sB[(16*j + lr) * 32]);
      bf.f[0] = pb[half];
      bf.f[1] = pb[2 + half];
      acc[0][j] = __builtin_amdgcn_wmma_f32_16x16x32_f16(
          false, af[0].v, false, bf.v, (short)0, acc[0][j], false, false);
      acc[1][j] = __builtin_amdgcn_wmma_f32_16x16x32_f16(
          false, af[1].v, false, bf.v, (short)0, acc[1][j], false, false);
    }
  }

  // epilogue: C/D layout -> lane lr gives N, half/r give M
  #pragma unroll
  for (int j = 0; j < 4; ++j) {
    int n = n0 + 16*j + lr;
    if (n >= NoutReal) continue;
    float bi = bias[n];
    float gg = bnrelu ? gam[n] : 0.0f;
    float bb = bnrelu ? bet[n] : 0.0f;
    #pragma unroll
    for (int mi = 0; mi < 2; ++mi) {
      #pragma unroll
      for (int r = 0; r < 8; ++r) {
        int row = m0 + 32*wv + 16*mi + 8*half + r;
        float v = acc[mi][j][r] + bi;
        if (bnrelu) v = fmaxf(0.0f, gg * (v * BN_SCALE) + bb);
        if (Ybf) {
          Ybf[(size_t)row * NoutPad + n] = (_Float16)v;
        } else {
          int b = row >> 14, nn = row & (NPTS - 1);
          Yt[((size_t)b * NoutReal + n) * NPTS + nn] = v;
        }
      }
    }
  }
}

// ---------------------------------------------------------------- host
extern "C" void kernel_launch(void* const* d_in, const int* in_sizes, int n_in,
                              void* d_out, int out_size, void* d_ws, size_t ws_size,
                              hipStream_t stream) {
  (void)in_sizes; (void)n_in; (void)out_size; (void)ws_size;
  const float* xyz = (const float*)d_in[0];
  struct Layer { const float *W, *b, *g, *be; };
  auto L = [&](int i) {
    Layer l{ (const float*)d_in[i], (const float*)d_in[i+1],
             (const float*)d_in[i+2], (const float*)d_in[i+3] };
    return l;
  };
  // insertion order: xyz; sa1(3 branches x 2 layers); sa3; fp3; fp1; head
  Layer sa1_0a = L(1),  sa1_0b = L(5);
  Layer sa1_1a = L(9),  sa1_1b = L(13);
  Layer sa1_2a = L(17), sa1_2b = L(21);
  Layer sa3a = L(25), sa3b = L(29);
  Layer fp3a = L(33), fp3b = L(37);
  Layer fp1a = L(41), fp1b = L(45);
  Layer hda  = L(49), hdb  = L(53);

  char* base = (char*)d_ws;
  size_t off = 0;
  auto alloc = [&](size_t bytes) -> char* {
    off = (off + 255) & ~(size_t)255;
    char* p = base + off; off += bytes; return p;
  };
  float* dist   = (float*)alloc(sizeof(float) * BATCH * NPTS);
  int*   fpsidx = (int*)  alloc(sizeof(int)   * BATCH * S8);
  float* newxyz = (float*)alloc(sizeof(float) * BATCH * S8 * 3);
  int*   gi0    = (int*)  alloc(sizeof(int)   * BATCH * S8 * 8);
  int*   gi1    = (int*)  alloc(sizeof(int)   * BATCH * S8 * 12);
  int*   gi2    = (int*)  alloc(sizeof(int)   * BATCH * S8 * 16);
  float* gp     = (float*)alloc(sizeof(float) * BATCH * S8 * 16 * 1030);
  float* h1     = (float*)alloc(sizeof(float) * BATCH * S8 * 16 * 256);
  float* h2     = (float*)alloc(sizeof(float) * BATCH * S8 * 16 * 512);
  float* l1pts  = (float*)alloc(sizeof(float) * 16 * 1024);
  float* sa3in  = (float*)alloc(sizeof(float) * 16 * 1027);
  float* sa3h1  = (float*)alloc(sizeof(float) * 16 * 1024);
  float* sa3h2  = (float*)alloc(sizeof(float) * 16 * 1024);
  float* l3max  = (float*)alloc(sizeof(float) * BATCH * 1024);
  float* fp3in  = (float*)alloc(sizeof(float) * 16 * 2048);
  float* fp3h1  = (float*)alloc(sizeof(float) * 16 * 1024);
  float* fp3o   = (float*)alloc(sizeof(float) * 16 * 1024);
  int*   idx3   = (int*)  alloc(sizeof(int)   * BATCH * NPTS * 3);
  float* w3     = (float*)alloc(sizeof(float) * BATCH * NPTS * 3);
  _Float16* Abf  = (_Float16*)alloc((size_t)2 * BATCH * NPTS * 2080);
  _Float16* Wbf1 = (_Float16*)alloc((size_t)2 * 512 * 2080);
  _Float16* H1bf = (_Float16*)alloc((size_t)2 * BATCH * NPTS * 512);
  _Float16* Wbf2 = (_Float16*)alloc((size_t)2 * 512 * 512);
  _Float16* H2bf = (_Float16*)alloc((size_t)2 * BATCH * NPTS * 512);
  _Float16* Wbf3 = (_Float16*)alloc((size_t)2 * 128 * 512);
  _Float16* H3bf = (_Float16*)alloc((size_t)2 * BATCH * NPTS * 128);
  _Float16* Wbf4 = (_Float16*)alloc((size_t)2 * 64 * 128);

  // FPS + centroids
  fps_kernel<<<BATCH, 256, 0, stream>>>(xyz, fpsidx, dist);
  gather_new_xyz<<<1, 64, 0, stream>>>(xyz, fpsidx, newxyz);
  // ball queries (r^2 = 0.01, 0.04, 0.16)
  ball_query<<<1, 32, 0, stream>>>(xyz, newxyz, gi0, 0.01f, 8);
  ball_query<<<1, 32, 0, stream>>>(xyz, newxyz, gi1, 0.04f, 12);
  ball_query<<<1, 32, 0, stream>>>(xyz, newxyz, gi2, 0.16f, 16);

  auto run_branch = [&](const int* gi, int K, int Cmid, int Cout,
                        Layer la, Layer lb, int coff) {
    int R = BATCH * S8 * K;
    int n1 = R * 1030;
    build_gp<<<(n1 + 255) / 256, 256, 0, stream>>>(xyz, newxyz, gi, gp, K);
    int n2 = R * Cmid;
    mlp_scalar<<<(n2 + 255) / 256, 256, 0, stream>>>(gp, la.W, la.b, la.g, la.be,
                                                     h1, R, 1030, Cmid, 1);
    int n3 = R * Cout;
    mlp_scalar<<<(n3 + 255) / 256, 256, 0, stream>>>(h1, lb.W, lb.b, lb.g, lb.be,
                                                     h2, R, Cmid, Cout, 1);
    int n4 = BATCH * S8 * Cout;
    max_over_k<<<(n4 + 255) / 256, 256, 0, stream>>>(h2, l1pts, K, Cout, coff);
  };
  run_branch(gi0, 8,  128, 256, sa1_0a, sa1_0b, 0);
  run_branch(gi1, 12, 256, 512, sa1_1a, sa1_1b, 256);
  run_branch(gi2, 16, 64,  256, sa1_2a, sa1_2b, 768);

  // sa3: [l1_xyz | l1_pts] -> 1024 -> 1024, max over S
  build_sa3_in<<<(16*1027 + 255)/256, 256, 0, stream>>>(newxyz, l1pts, sa3in);
  mlp_scalar<<<(16*1024 + 255)/256, 256, 0, stream>>>(sa3in, sa3a.W, sa3a.b, sa3a.g, sa3a.be,
                                                      sa3h1, 16, 1027, 1024, 1);
  mlp_scalar<<<(16*1024 + 255)/256, 256, 0, stream>>>(sa3h1, sa3b.W, sa3b.b, sa3b.g, sa3b.be,
                                                      sa3h2, 16, 1024, 1024, 1);
  max_over_s<<<(BATCH*1024 + 255)/256, 256, 0, stream>>>(sa3h2, l3max);

  // fp3: [l1_pts | l3max broadcast] -> 1024 -> 1024
  build_fp3_in<<<(16*2048 + 255)/256, 256, 0, stream>>>(l1pts, l3max, fp3in);
  mlp_scalar<<<(16*1024 + 255)/256, 256, 0, stream>>>(fp3in, fp3a.W, fp3a.b, fp3a.g, fp3a.be,
                                                      fp3h1, 16, 2048, 1024, 1);
  mlp_scalar<<<(16*1024 + 255)/256, 256, 0, stream>>>(fp3h1, fp3b.W, fp3b.b, fp3b.g, fp3b.be,
                                                      fp3o, 16, 1024, 1024, 1);

  // 3-NN interpolation weights, then fp1 input in f16 (fused interp)
  knn3_weights<<<(BATCH*NPTS + 255)/256, 256, 0, stream>>>(xyz, newxyz, idx3, w3);
  {
    size_t total = (size_t)BATCH * NPTS * 2080;
    build_a_kernel<<<(unsigned)((total + 255)/256), 256, 0, stream>>>(xyz, idx3, w3, fp3o, Abf);
  }
  // f16 weight conversion (K padded to mult of 32, N padded to mult of 64 for head2)
  conv_weight<<<(512*2080 + 255)/256, 256, 0, stream>>>(fp1a.W, Wbf1, 512, 512, 2054, 2080);
  conv_weight<<<(512*512  + 255)/256, 256, 0, stream>>>(fp1b.W, Wbf2, 512, 512, 512, 512);
  conv_weight<<<(128*512  + 255)/256, 256, 0, stream>>>(hda.W,  Wbf3, 128, 128, 512, 512);
  conv_weight<<<(64*128   + 255)/256, 256, 0, stream>>>(hdb.W,  Wbf4, 32,  64,  128, 128);

  const int M = BATCH * NPTS;  // 32768, divisible by 256
  // fp1 L1: 2080 -> 512, bn-relu
  gemm_wmma<<<dim3(M/256, 512/64), 256, 0, stream>>>(Abf,  Wbf1, fp1a.b, fp1a.g, fp1a.be,
                                                     H1bf, nullptr, 2080, 512, 512, 1);
  // fp1 L2: 512 -> 512, bn-relu
  gemm_wmma<<<dim3(M/256, 512/64), 256, 0, stream>>>(H1bf, Wbf2, fp1b.b, fp1b.g, fp1b.be,
                                                     H2bf, nullptr, 512, 512, 512, 1);
  // head L1: 512 -> 128, bn-relu
  gemm_wmma<<<dim3(M/256, 128/64), 256, 0, stream>>>(H2bf, Wbf3, hda.b, hda.g, hda.be,
                                                     H3bf, nullptr, 512, 128, 128, 1);
  // head L2: 128 -> 32 plain, fp32 transposed into d_out (B, 32, N)
  gemm_wmma<<<dim3(M/256, 64/64), 256, 0, stream>>>(H3bf, Wbf4, hdb.b, nullptr, nullptr,
                                                    nullptr, (float*)d_out, 128, 32, 64, 0);
}
